// AttentionLayer_89455578841810
// MI455X (gfx1250) — matrix operand order repaired and checked by reading, not compiled, not run on hardware
//
#include <hip/hip_runtime.h>
#include <hip/hip_bf16.h>
#include <stdint.h>

// Problem constants (from reference)
#define BB 2
#define LL 2048
#define SS 2048
#define DD 512
#define HH 8
#define DKK 64
#define HD  512   // H*DK

typedef __bf16 v16bf __attribute__((ext_vector_type(16)));
typedef float  v8f   __attribute__((ext_vector_type(8)));
typedef unsigned int u32x4 __attribute__((ext_vector_type(4)));
typedef int          i32x4 __attribute__((ext_vector_type(4)));
typedef int          i32x8 __attribute__((ext_vector_type(8)));

// ---------- helpers ----------
static __device__ __forceinline__ __bf16 f2bf(float f) {
    unsigned u; __builtin_memcpy(&u, &f, 4);
    unsigned r = (u + 0x7FFFu + ((u >> 16) & 1u)) >> 16;   // round-to-nearest-even
    unsigned short s = (unsigned short)r;
    __bf16 o; __builtin_memcpy(&o, &s, 2);
    return o;
}
static __device__ __forceinline__ float bf2f(__bf16 x) {
    unsigned short s; __builtin_memcpy(&s, &x, 2);
    unsigned u = ((unsigned)s) << 16;
    float f; __builtin_memcpy(&f, &u, 4);
    return f;
}
static __device__ __forceinline__ v8f wmma_bf16(v16bf a, v16bf b, v8f c) {
    return __builtin_amdgcn_wmma_f32_16x16x32_bf16(false, a, false, b, (short)0, c, false, false);
}

// ---------- Tensor Data Mover staging (CDNA5) ----------
#if defined(__AMDGCN__) && __has_builtin(__builtin_amdgcn_tensor_load_to_lds) && __has_builtin(__builtin_amdgcn_s_wait_tensorcnt)
#define USE_TDM 1
#else
#define USE_TDM 0
#endif

#if USE_TDM
// 2D tile (tileRows x tileCols elements) from a row-major matrix into LDS (packed).
// dataSizeCode: 0=1B, 1=2B (bf16), 2=4B (f32), 3=8B  (D# group1 data_size field)
static __device__ __forceinline__ void tdm_load_2d(void* lds, const void* g,
                                                   unsigned tileCols, unsigned tileRows,
                                                   unsigned rowStrideElems, unsigned dataSizeCode) {
    unsigned long long ga = (unsigned long long)(uintptr_t)g;
    unsigned ldsOff = (unsigned)(uintptr_t)lds;            // low 32 bits of generic ptr == LDS byte offset
    u32x4 g0; i32x8 g1; i32x4 g2, g3;
    g0[0] = 1u;                                            // count = 1 valid descriptor
    g0[1] = ldsOff;                                        // lds_addr
    g0[2] = (unsigned)(ga & 0xffffffffull);                // global_addr lo
    g0[3] = (unsigned)((ga >> 32) & 0x01ffffffull) | (2u << 30); // global_addr hi | type=2 (image)
    g1[0] = (int)(dataSizeCode << 16);                     // data_size
    g1[1] = (int)((rowStrideElems & 0xffffu) << 16);       // tensor_dim0[15:0] @ [31:16]
    g1[2] = (int)((rowStrideElems >> 16) | ((tileRows & 0xffffu) << 16)); // dim0 hi | dim1 lo
    g1[3] = (int)((tileRows >> 16) | ((tileCols & 0xffffu) << 16));       // dim1 hi | tile_dim0
    g1[4] = (int)(tileRows & 0xffffu);                     // tile_dim1 (tile_dim2 = 0)
    g1[5] = (int)rowStrideElems;                           // tensor_dim0_stride lo32
    g1[6] = 0; g1[7] = 0;
    g2[0] = g2[1] = g2[2] = g2[3] = 0;
    g3[0] = g3[1] = g3[2] = g3[3] = 0;
#if __clang_major__ >= 23
    i32x8 g4; g4[0]=g4[1]=g4[2]=g4[3]=g4[4]=g4[5]=g4[6]=g4[7]=0;
    __builtin_amdgcn_tensor_load_to_lds(g0, g1, g2, g3, g4, 0);
#else
    __builtin_amdgcn_tensor_load_to_lds(g0, g1, g2, g3, 0);
#endif
    __builtin_amdgcn_s_wait_tensorcnt(0);
}
#endif

// =====================================================================
// Kernel 1/5: GEMM C[M=4096, 512] = A[4096,512] x W[512,512] + bias
// 4 waves/WG, each wave does a 16x64 strip (4 WMMA column tiles), K-loop by 32.
// W tiles staged through LDS via TDM when available.
// =====================================================================
template <bool A_BF16, bool OUT_F32>
__global__ __launch_bounds__(128) void gemm512_kernel(const void* __restrict__ Ap,
                                                      const float* __restrict__ W,
                                                      const float* __restrict__ bias,
                                                      void* __restrict__ Outp) {
    __shared__ float smemW[32 * 64];
    const int row0 = blockIdx.x * 64;
    const int col0 = blockIdx.y * 64;
    const int tid  = threadIdx.x;
    const int lane = tid & 31;
    const int wave = tid >> 5;
    const int r0   = row0 + wave * 16;
    const int m    = lane & 15;
    const int hi   = lane >> 4;
    const int n    = lane & 15;
    const int kb   = hi * 8;     // A fragment K group base
    const int kb2  = hi * 16;    // unused (B fragment also uses kb)

    v8f acc[4];
    #pragma unroll
    for (int t = 0; t < 4; ++t)
        #pragma unroll
        for (int r = 0; r < 8; ++r) acc[t][r] = 0.0f;

    for (int k0 = 0; k0 < 512; k0 += 32) {
        __syncthreads();
#if USE_TDM
        if (tid < 32) tdm_load_2d(smemW, W + (size_t)k0 * 512 + col0, 64u, 32u, 512u, 2u);
#else
        for (int i = tid; i < 32 * 64; i += 128) {
            int kk = i >> 6, nn = i & 63;
            smemW[i] = W[(size_t)(k0 + kk) * 512 + col0 + nn];
        }
#endif
        __syncthreads();

        v16bf a;
        if constexpr (A_BF16) {
            const __bf16* Arow = (const __bf16*)Ap + (size_t)(r0 + m) * 512 + k0;
            #pragma unroll
            for (int i = 0; i < 8; ++i) { a[i] = Arow[kb + i]; a[8 + i] = Arow[kb + 16 + i]; }
        } else {
            const float* Arow = (const float*)Ap + (size_t)(r0 + m) * 512 + k0;
            __builtin_prefetch(Arow + 32, 0, 1);           // global_prefetch_b8 for next K step
            #pragma unroll
            for (int i = 0; i < 8; ++i) { a[i] = f2bf(Arow[kb + i]); a[8 + i] = f2bf(Arow[kb + 16 + i]); }
        }

        #pragma unroll
        for (int t = 0; t < 4; ++t) {
            v16bf bfr;
            #pragma unroll
            for (int j = 0; j < 16; ++j) bfr[j] = f2bf(smemW[(kb + j + (j & 8)) * 64 + t * 16 + n]);
            acc[t] = wmma_bf16(a, bfr, acc[t]);
        }
    }

    #pragma unroll
    for (int t = 0; t < 4; ++t) {
        #pragma unroll
        for (int r = 0; r < 8; ++r) {
            int mm  = r + 8 * hi;
            int col = col0 + t * 16 + n;
            float val = acc[t][r] + bias[col];
            if constexpr (OUT_F32)
                ((float*)Outp)[(size_t)(r0 + mm) * 512 + col] = val;
            else
                ((__bf16*)Outp)[(size_t)(r0 + mm) * 512 + col] = f2bf(val);
        }
    }
}

// =====================================================================
// Kernel 2/5: in-place RoPE on bf16 [B, L, H, DK]
// =====================================================================
__global__ __launch_bounds__(256) void rope_bf16_kernel(__bf16* __restrict__ x,
                                                        const float* __restrict__ cosT,
                                                        const float* __restrict__ sinT) {
    int i = blockIdx.x * blockDim.x + threadIdx.x;
    if (i >= BB * LL * HH * (DKK / 2)) return;
    int j = i & 31;                 // DK/2 = 32
    int h = (i >> 5) & 7;           // H = 8
    int l = (i >> 8) & 2047;        // L = 2048
    int b = i >> 19;
    size_t base = (((size_t)b * LL + l) * HH + h) * DKK;
    float x0 = bf2f(x[base + 2 * j]);
    float x1 = bf2f(x[base + 2 * j + 1]);
    float c = cosT[l * 32 + j];
    float s = sinT[l * 32 + j];
    x[base + 2 * j]     = f2bf(x0 * c - x1 * s);
    x[base + 2 * j + 1] = f2bf(x0 * s + x1 * c);
}

// =====================================================================
// Kernel 3/5: scores = (q k^T)/8 for a 16-row block, softmax fused in LDS.
// 8 waves; each wave covers 16 column tiles of 16; 16x2048 fp32 slab in
// dynamic LDS (128 KB — fits CDNA5's 320 KB/WGP). Writes fp32 attn.
// =====================================================================
__global__ __launch_bounds__(256) void scores_softmax_kernel(const __bf16* __restrict__ q,
                                                             const __bf16* __restrict__ k,
                                                             float* __restrict__ attn) {
    extern __shared__ float sc[];   // [16][2048]
    const int l0 = blockIdx.x * 16;
    const int h  = blockIdx.y;
    const int b  = blockIdx.z;
    const int tid  = threadIdx.x;
    const int lane = tid & 31;
    const int wave = tid >> 5;      // 0..7
    const int m  = lane & 15;
    const int hi = lane >> 4;
    const int n  = lane & 15;
    const int kb  = hi * 8;
    const int kb2 = hi * 16;

    // A fragments for K steps 0 and 32 (DK = 64)
    v16bf a0, a1;
    {
        const __bf16* qrow = q + (((size_t)b * LL + l0 + m) * HH + h) * DKK;
        #pragma unroll
        for (int i = 0; i < 8; ++i) {
            a0[i]     = qrow[kb + i];
            a0[8 + i] = qrow[kb + 16 + i];
            a1[i]     = qrow[32 + kb + i];
            a1[8 + i] = qrow[32 + kb + 16 + i];
        }
    }

    #pragma unroll 4
    for (int t = 0; t < 16; ++t) {
        int s0 = (wave * 16 + t) * 16;
        const __bf16* krow = k + (((size_t)b * SS + s0 + n) * HH + h) * DKK;
        v16bf b0, b1;
        #pragma unroll
        for (int j = 0; j < 16; ++j) {
            b0[j] = krow[kb + j + (j & 8)];
            b1[j] = krow[32 + kb + j + (j & 8)];
        }
        v8f acc;
        #pragma unroll
        for (int r = 0; r < 8; ++r) acc[r] = 0.0f;
        acc = wmma_bf16(a0, b0, acc);
        acc = wmma_bf16(a1, b1, acc);
        #pragma unroll
        for (int r = 0; r < 8; ++r)
            sc[(r + 8 * hi) * SS + s0 + n] = acc[r] * 0.125f;   // 1/sqrt(64)
    }
    __syncthreads();

    // Row softmax: 16 rows, 16 threads per row; shuffle reduce within 16-lane groups
    const int row = tid >> 4;
    const int sub = tid & 15;
    const float* srow = &sc[row * SS];
    float mx = -3.0e38f;
    for (int c = sub; c < SS; c += 16) mx = fmaxf(mx, srow[c]);
    #pragma unroll
    for (int o = 1; o < 16; o <<= 1) mx = fmaxf(mx, __shfl_xor(mx, o, 16));
    float sum = 0.0f;
    for (int c = sub; c < SS; c += 16) sum += __expf(srow[c] - mx);
    #pragma unroll
    for (int o = 1; o < 16; o <<= 1) sum += __shfl_xor(sum, o, 16);
    float inv = 1.0f / sum;
    float* arow = attn + (((size_t)b * HH + h) * LL + l0 + row) * SS;
    for (int c = sub; c < SS; c += 16) arow[c] = __expf(srow[c] - mx) * inv;
}

// =====================================================================
// Kernel 4/5: out_h = attn @ v   (M=16 rows/WG, N=64, K=2048)
// 4 waves, one 16-col tile each; attn fp32 converted to bf16 on load.
// V tile (32 x 64 bf16) staged into LDS via TDM so the per-lane fragment
// gathers hit LDS instead of 512-element-strided global locations.
// =====================================================================
__global__ __launch_bounds__(128) void av_gemm_kernel(const float* __restrict__ attn,
                                                      const __bf16* __restrict__ v,
                                                      __bf16* __restrict__ outh) {
    __shared__ __bf16 smemV[32 * 64];
    const int l0 = blockIdx.x * 16;
    const int h  = blockIdx.y;
    const int b  = blockIdx.z;
    const int tid  = threadIdx.x;
    const int lane = tid & 31;
    const int wave = tid >> 5;      // 0..3
    const int n0 = wave * 16;
    const int m  = lane & 15;
    const int hi = lane >> 4;
    const int n  = lane & 15;
    const int kb  = hi * 8;
    const int kb2 = hi * 16;

    v8f acc;
    #pragma unroll
    for (int r = 0; r < 8; ++r) acc[r] = 0.0f;

    const float* arow = attn + (((size_t)b * HH + h) * LL + l0 + m) * SS;
    for (int s0 = 0; s0 < SS; s0 += 32) {
        __syncthreads();
#if USE_TDM
        if (tid < 32)
            tdm_load_2d(smemV, v + (((size_t)b * SS + s0) * HH + h) * DKK, 64u, 32u, 512u, 1u);
#else
        for (int i = tid; i < 32 * 64; i += 128) {
            int ss = i >> 6, dd = i & 63;
            smemV[i] = v[(((size_t)b * SS + s0 + ss) * HH + h) * DKK + dd];
        }
#endif
        __syncthreads();

        v16bf a, bfr;
        #pragma unroll
        for (int i = 0; i < 8; ++i) {
            a[i]     = f2bf(arow[s0 + kb + i]);
            a[8 + i] = f2bf(arow[s0 + kb + 16 + i]);
        }
        #pragma unroll
        for (int j = 0; j < 16; ++j)
            bfr[j] = smemV[(kb + j + (j & 8)) * 64 + n0 + n];
        acc = wmma_bf16(a, bfr, acc);
    }
    #pragma unroll
    for (int r = 0; r < 8; ++r) {
        int mm = r + 8 * hi;
        outh[(((size_t)b * LL + l0 + mm) * HH + h) * DKK + n0 + n] = f2bf(acc[r]);
    }
}

// =====================================================================
extern "C" void kernel_launch(void* const* d_in, const int* in_sizes, int n_in,
                              void* d_out, int out_size, void* d_ws, size_t ws_size,
                              hipStream_t stream) {
    const float* queries = (const float*)d_in[0];
    const float* keys    = (const float*)d_in[1];
    const float* values  = (const float*)d_in[2];
    const float* fcos    = (const float*)d_in[3];
    const float* fsin    = (const float*)d_in[4];
    const float* Wq = (const float*)d_in[5];  const float* bq = (const float*)d_in[6];
    const float* Wk = (const float*)d_in[7];  const float* bk = (const float*)d_in[8];
    const float* Wv = (const float*)d_in[9];  const float* bv = (const float*)d_in[10];
    const float* Wo = (const float*)d_in[11]; const float* bo = (const float*)d_in[12];

    // Workspace: bf16 q, k, v, attn_out  (each B*L*H*DK = 2,097,152 elems = 4 MB)
    const size_t NE = (size_t)BB * LL * HH * DKK;
    __bf16* qb = (__bf16*)d_ws;
    __bf16* kb = qb + NE;
    __bf16* vb = kb + NE;
    __bf16* ob = vb + NE;

    float* outp = (float*)d_out;                 // [B, L, D]
    float* attn = outp + (size_t)BB * LL * DD;   // [B, H, L, S]

    dim3 gGemm(4096 / 64, 512 / 64);             // M/64 x N/64
    // QKV projections (fp32 A/W -> bf16 compute -> bf16 out)
    gemm512_kernel<false, false><<<gGemm, 128, 0, stream>>>(queries, Wq, bq, qb);
    gemm512_kernel<false, false><<<gGemm, 128, 0, stream>>>(keys,    Wk, bk, kb);
    gemm512_kernel<false, false><<<gGemm, 128, 0, stream>>>(values,  Wv, bv, vb);

    // RoPE on q, k
    int ropeN = BB * LL * HH * (DKK / 2);
    rope_bf16_kernel<<<(ropeN + 255) / 256, 256, 0, stream>>>(qb, fcos, fsin);
    rope_bf16_kernel<<<(ropeN + 255) / 256, 256, 0, stream>>>(kb, fcos, fsin);

    // scores + fused softmax -> attn (fp32, second output)
    dim3 gAttn(LL / 16, HH, BB);
    scores_softmax_kernel<<<gAttn, 256, 16 * SS * sizeof(float), stream>>>(qb, kb, attn);

    // attn @ v -> bf16 head output
    av_gemm_kernel<<<gAttn, 128, 0, stream>>>(attn, vb, ob);

    // output projection (bf16 A -> fp32 out)
    gemm512_kernel<true, true><<<gGemm, 128, 0, stream>>>(ob, Wo, bo, outp);
}
